// AttentionBlock_78400333021395
// MI455X (gfx1250) — matrix operand order repaired and checked by reading, not compiled, not run on hardware
//
#include <hip/hip_runtime.h>
#include <hip/hip_bf16.h>
#include <math.h>

typedef __attribute__((ext_vector_type(16))) _Float16 v16h;
typedef __attribute__((ext_vector_type(8)))  float    v8f;

#define BSZ     64
#define CDIM    512
#define HWN     196
#define NT      224          // tokens padded to 14*16
#define NH      16
#define HD      32
#define HID     2048
#define EPSBN   1e-5f
#define SCALE   0.1767766952966369f   // 32^-0.5

union FragH { v16h h; unsigned u[8]; };

// ---------------------------------------------------------------------------
// weight f32 -> f16 convert
__global__ void cvt_f16_kernel(const float* __restrict__ src,
                               _Float16* __restrict__ dst, int n) {
  int i = blockIdx.x * blockDim.x + threadIdx.x;
  if (i < n) dst[i] = (_Float16)src[i];
}

// ---------------------------------------------------------------------------
// BN(x) (NCHW) -> f16 token-major [B][NT][C], padded rows zero
__global__ void pack_bn_kernel(const float* __restrict__ x,
                               const float* __restrict__ g, const float* __restrict__ bch,
                               const float* __restrict__ m, const float* __restrict__ v,
                               _Float16* __restrict__ dst) {
  long i = (long)blockIdx.x * blockDim.x + threadIdx.x;
  long total = (long)BSZ * NT * CDIM;
  if (i >= total) return;
  int c = (int)(i % CDIM);
  long bn = i / CDIM;
  int n = (int)(bn % NT);
  int b = (int)(bn / NT);
  float val = 0.f;
  if (n < HWN) {
    float inv = g[c] * rsqrtf(v[c] + EPSBN);
    float beta = bch[c] - m[c] * inv;
    val = x[((long)b * CDIM + c) * HWN + n] * inv + beta;
  }
  dst[i] = (_Float16)val;
}

// ---------------------------------------------------------------------------
// Generic WMMA GEMM:  D[m,n] = sum_k A[m,k] * W[n,k]   (A,W f16 row-major)
// grid: (Nout/512, NT/32, B)  block: 256 threads = 8 waves
// block tile 32(M) x 512(N); wave tile 32(M) x 64(N): 2 A-frags, 4 B-frags,
// 8 WMMAs per k-step (1.5 B128 loads per WMMA, 8 independent WMMA chains)
// EPI 0: f16 store   EPI 1: f32 store (+bias)   EPI 2: f16 GELU(+bias)
template <int EPI>
__global__ __launch_bounds__(256) void gemm_wmma_kernel(
    const _Float16* __restrict__ A, const _Float16* __restrict__ W,
    const float* __restrict__ bias, void* __restrict__ out,
    int K, int Nout) {
  const int lane = threadIdx.x & 31;
  const int wave = threadIdx.x >> 5;
  const int m0 = blockIdx.y * 32;
  const int n0 = blockIdx.x * 512 + wave * 64;
  const long abase = (long)blockIdx.z * NT * K;
  const long obase = (long)blockIdx.z * NT * Nout;

  const int arow0 = m0 + (lane & 15);
  const int akoff = (lane >> 4) << 4;   // +0 or +16
  const int bkoff = (lane >> 4) << 4;   // +0 or +16

  v8f acc[2][4] = {};
  for (int k0 = 0; k0 < K; k0 += 32) {
    FragH a[2];
#pragma unroll
    for (int mi = 0; mi < 2; ++mi) {
      const unsigned* pa =
          (const unsigned*)(A + abase + (long)(arow0 + 16 * mi) * K + k0 + akoff);
      a[mi].u[0] = pa[0]; a[mi].u[1] = pa[1]; a[mi].u[2] = pa[2]; a[mi].u[3] = pa[3];
      a[mi].u[4] = pa[4]; a[mi].u[5] = pa[5]; a[mi].u[6] = pa[6]; a[mi].u[7] = pa[7];
    }
#pragma unroll
    for (int t = 0; t < 4; ++t) {
      FragH bf;
      const int col = n0 + t * 16 + (lane & 15);
      const unsigned* pb = (const unsigned*)(W + (long)col * K + k0 + bkoff);
#pragma unroll
      for (int j = 0; j < 8; ++j) bf.u[j] = pb[j];
#pragma unroll
      for (int mi = 0; mi < 2; ++mi)
        acc[mi][t] = __builtin_amdgcn_wmma_f32_16x16x32_f16(
            false, a[mi].h, false, bf.h, (short)0, acc[mi][t], false, false);
    }
  }
  // epilogue
#pragma unroll
  for (int mi = 0; mi < 2; ++mi) {
#pragma unroll
    for (int t = 0; t < 4; ++t) {
      const int col = n0 + t * 16 + (lane & 15);
      const float bb = (EPI == 0) ? 0.f : bias[col];
#pragma unroll
      for (int r = 0; r < 8; ++r) {
        const int row = m0 + 16 * mi + r + ((lane >> 4) << 3);
        float val = acc[mi][t][r] + bb;
        long idx = obase + (long)row * Nout + col;
        if (EPI == 0) {
          ((_Float16*)out)[idx] = (_Float16)val;
        } else if (EPI == 1) {
          ((float*)out)[idx] = val;
        } else {
          float gl = 0.5f * val * (1.0f + erff(val * 0.7071067811865475f));
          ((_Float16*)out)[idx] = (_Float16)gl;
        }
      }
    }
  }
}

// ---------------------------------------------------------------------------
// transpose V out of qkv into Vt[b][h][d][key]  (f16, key contiguous)
__global__ void vt_pack_kernel(const _Float16* __restrict__ qkv,
                               _Float16* __restrict__ vt) {
  long i = (long)blockIdx.x * blockDim.x + threadIdx.x;
  long total = (long)BSZ * NH * HD * NT;
  if (i >= total) return;
  int k = (int)(i % NT);
  long rem = i / NT;
  int d = (int)(rem % HD);
  long bh = rem / HD;
  int h = (int)(bh % NH);
  int b = (int)(bh / NH);
  vt[i] = qkv[((long)b * NT + k) * 1536 + 1024 + h * 32 + d];
}

// ---------------------------------------------------------------------------
// fused attention: one wave per (b, head, 16-row query tile)
// qkv f16 [B][NT][1536] col = sel*512+h*32+d ; Vt f16 [B][NH][32][NT]
// O f16 [B][NT][512]
__global__ __launch_bounds__(32) void attn_kernel(const _Float16* __restrict__ qkv,
                                                  const _Float16* __restrict__ vt,
                                                  _Float16* __restrict__ O) {
  __shared__ float    sS[16 * NT];
  __shared__ _Float16 sP[16 * NT];
  const int lane = threadIdx.x;
  const int m0 = blockIdx.x * 16;
  const int h  = blockIdx.y;
  const int b  = blockIdx.z;
  const long base = (long)b * NT * 1536;
  const int hi = (lane >> 4);          // 0/1 half of wave

  // Q fragment (16x32), single K step (HD == 32)
  FragH qa;
  {
    const int row = m0 + (lane & 15);
    const unsigned* pa =
        (const unsigned*)(qkv + base + (long)row * 1536 + h * 32 + (hi << 4));
    qa.u[0] = pa[0]; qa.u[1] = pa[1]; qa.u[2] = pa[2]; qa.u[3] = pa[3];
    qa.u[4] = pa[4]; qa.u[5] = pa[5]; qa.u[6] = pa[6]; qa.u[7] = pa[7];
  }
  // S = Q @ K^T over all 14 key tiles
  for (int nt = 0; nt < NT / 16; ++nt) {
    FragH kb;
    const int key = nt * 16 + (lane & 15);
    const unsigned* pb =
        (const unsigned*)(qkv + base + (long)key * 1536 + 512 + h * 32 + (hi << 4));
#pragma unroll
    for (int j = 0; j < 8; ++j) kb.u[j] = pb[j];
    v8f s = {};
    s = __builtin_amdgcn_wmma_f32_16x16x32_f16(false, qa.h, false, kb.h,
                                               (short)0, s, false, false);
#pragma unroll
    for (int r = 0; r < 8; ++r) {
      const int mrow = r + (hi << 3);
      sS[mrow * NT + nt * 16 + (lane & 15)] = s[r] * SCALE;
    }
  }
  __syncthreads();
  // masked softmax, one row per lane (lanes 0..15)
  if (lane < 16) {
    const float* row = &sS[lane * NT];
    float mx = -3.0e38f;
    for (int c = 0; c < HWN; ++c) mx = fmaxf(mx, row[c]);
    float sum = 0.f;
    for (int c = 0; c < HWN; ++c) sum += expf(row[c] - mx);
    const float rs = 1.0f / sum;
    _Float16* prow = &sP[lane * NT];
    for (int c = 0; c < NT; ++c)
      prow[c] = (c < HWN) ? (_Float16)(expf(row[c] - mx) * rs) : (_Float16)0.0f;
  }
  __syncthreads();
  // O = P @ V  (K = NT keys, 7 steps; two 16-wide d tiles; V from Vt, B128)
  const _Float16* vth = vt + ((long)b * NH + h) * HD * NT;
  v8f o0 = {}, o1 = {};
  for (int k0 = 0; k0 < NT; k0 += 32) {
    FragH pa;
    {
      const int row = lane & 15;
      const unsigned* pp = (const unsigned*)(&sP[row * NT + k0 + (hi << 4)]);
      pa.u[0] = pp[0]; pa.u[1] = pp[1]; pa.u[2] = pp[2]; pa.u[3] = pp[3];
      pa.u[4] = pp[4]; pa.u[5] = pp[5]; pa.u[6] = pp[6]; pa.u[7] = pp[7];
    }
#pragma unroll
    for (int t = 0; t < 2; ++t) {
      FragH vb;
      const int d = t * 16 + (lane & 15);
      const unsigned* pv = (const unsigned*)(vth + (long)d * NT + k0 + (hi << 4));
#pragma unroll
      for (int j = 0; j < 8; ++j) vb.u[j] = pv[j];
      v8f& acc = t ? o1 : o0;
      acc = __builtin_amdgcn_wmma_f32_16x16x32_f16(false, pa.h, false, vb.h,
                                                   (short)0, acc, false, false);
    }
  }
#pragma unroll
  for (int t = 0; t < 2; ++t) {
    const v8f& acc = t ? o1 : o0;
    const int d = t * 16 + (lane & 15);
#pragma unroll
    for (int r = 0; r < 8; ++r) {
      const int row = m0 + r + (hi << 3);
      O[((long)b * NT + row) * CDIM + h * 32 + d] = (_Float16)acc[r];
    }
  }
}

// ---------------------------------------------------------------------------
// x1 = x + ls1 * proj_out   (proj_out f32 token-major, x NCHW)
__global__ void residual1_kernel(const float* __restrict__ x,
                                 const float* __restrict__ proj,
                                 const float* __restrict__ ls1,
                                 float* __restrict__ x1) {
  long i = (long)blockIdx.x * blockDim.x + threadIdx.x;
  long total = (long)BSZ * CDIM * HWN;
  if (i >= total) return;
  int n = (int)(i % HWN);
  long bc = i / HWN;
  int c = (int)(bc % CDIM);
  int b = (int)(bc / CDIM);
  x1[i] = x[i] + ls1[c] * proj[((long)b * NT + n) * CDIM + c];
}

// ---------------------------------------------------------------------------
// BN(x1) -> depthwise 7x7 -> fBN -> f16 token-major [B][NT][C]
__global__ void dwconv_kernel(const float* __restrict__ x1,
                              const float* __restrict__ g, const float* __restrict__ bch,
                              const float* __restrict__ m, const float* __restrict__ v,
                              const float* __restrict__ dw,
                              const float* __restrict__ fg, const float* __restrict__ fb,
                              const float* __restrict__ fm, const float* __restrict__ fv,
                              _Float16* __restrict__ dst) {
  long i = (long)blockIdx.x * blockDim.x + threadIdx.x;
  long total = (long)BSZ * NT * CDIM;
  if (i >= total) return;
  int c = (int)(i % CDIM);
  long bn = i / CDIM;
  int n = (int)(bn % NT);
  int b = (int)(bn / NT);
  if (n >= HWN) { dst[i] = (_Float16)0.0f; return; }
  const int hh = n / 14, ww = n % 14;
  const float inv = g[c] * rsqrtf(v[c] + EPSBN);
  const float beta = bch[c] - m[c] * inv;
  const float* xc = x1 + ((long)b * CDIM + c) * HWN;
  const float* wc = dw + c * 49;
  float acc = 0.f;
#pragma unroll
  for (int ky = 0; ky < 7; ++ky) {
    const int iy = hh + ky - 3;
    if (iy < 0 || iy >= 14) continue;
#pragma unroll
    for (int kx = 0; kx < 7; ++kx) {
      const int ix = ww + kx - 3;
      if (ix < 0 || ix >= 14) continue;
      acc += wc[ky * 7 + kx] * (xc[iy * 14 + ix] * inv + beta);
    }
  }
  const float finv = fg[c] * rsqrtf(fv[c] + EPSBN);
  const float fbeta = fb[c] - fm[c] * finv;
  dst[i] = (_Float16)(acc * finv + fbeta);
}

// ---------------------------------------------------------------------------
// out = x1 + ls2 * fc2_out
__global__ void final_kernel(const float* __restrict__ x1,
                             const float* __restrict__ y,
                             const float* __restrict__ ls2,
                             float* __restrict__ out) {
  long i = (long)blockIdx.x * blockDim.x + threadIdx.x;
  long total = (long)BSZ * CDIM * HWN;
  if (i >= total) return;
  int n = (int)(i % HWN);
  long bc = i / HWN;
  int c = (int)(bc % CDIM);
  int b = (int)(bc / CDIM);
  out[i] = x1[i] + ls2[c] * y[((long)b * NT + n) * CDIM + c];
}

// ---------------------------------------------------------------------------
extern "C" void kernel_launch(void* const* d_in, const int* in_sizes, int n_in,
                              void* d_out, int out_size, void* d_ws, size_t ws_size,
                              hipStream_t stream) {
  const float* x      = (const float*)d_in[0];
  const float* bn_g   = (const float*)d_in[1];
  const float* bn_b   = (const float*)d_in[2];
  const float* bn_m   = (const float*)d_in[3];
  const float* bn_v   = (const float*)d_in[4];
  const float* qkv_w  = (const float*)d_in[5];
  const float* proj_w = (const float*)d_in[6];
  const float* proj_b = (const float*)d_in[7];
  const float* dw_w   = (const float*)d_in[8];
  const float* fbn_g  = (const float*)d_in[9];
  const float* fbn_b  = (const float*)d_in[10];
  const float* fbn_m  = (const float*)d_in[11];
  const float* fbn_v  = (const float*)d_in[12];
  const float* fc1_w  = (const float*)d_in[13];
  const float* fc1_b  = (const float*)d_in[14];
  const float* fc2_w  = (const float*)d_in[15];
  const float* fc2_b  = (const float*)d_in[16];
  const float* ls1    = (const float*)d_in[17];
  const float* ls2    = (const float*)d_in[18];

  // workspace carve-up (aliasing across pipeline phases)
  char* p = (char*)d_ws;
  auto take = [&](size_t bytes) { char* r = p; p += (bytes + 255) & ~size_t(255); return r; };
  _Float16* apack  = (_Float16*)take((size_t)BSZ * NT * CDIM * 2);   // also ffn pack
  _Float16* qkv16  = (_Float16*)take((size_t)BSZ * NT * HID * 2);    // qkv, later fc1 out
  _Float16* o16    = (_Float16*)take((size_t)BSZ * NT * CDIM * 2);
  _Float16* vt16   = (_Float16*)take((size_t)BSZ * NH * HD * NT * 2);
  float*    pout   = (float*)   take((size_t)BSZ * NT * CDIM * 4);   // proj, later fc2 out
  float*    x1     = (float*)   take((size_t)BSZ * CDIM * HWN * 4);
  _Float16* qkvw16 = (_Float16*)take((size_t)3 * CDIM * CDIM * 2);
  _Float16* projw16= (_Float16*)take((size_t)CDIM * CDIM * 2);
  _Float16* fc1w16 = (_Float16*)take((size_t)HID * CDIM * 2);
  _Float16* fc2w16 = (_Float16*)take((size_t)CDIM * HID * 2);

  const int TPB = 256;
  auto blocks = [&](long n) { return (unsigned)((n + TPB - 1) / TPB); };

  // 1) weights to f16
  cvt_f16_kernel<<<blocks(3L * CDIM * CDIM), TPB, 0, stream>>>(qkv_w, qkvw16, 3 * CDIM * CDIM);
  cvt_f16_kernel<<<blocks((long)CDIM * CDIM), TPB, 0, stream>>>(proj_w, projw16, CDIM * CDIM);
  cvt_f16_kernel<<<blocks((long)HID * CDIM), TPB, 0, stream>>>(fc1_w, fc1w16, HID * CDIM);
  cvt_f16_kernel<<<blocks((long)CDIM * HID), TPB, 0, stream>>>(fc2_w, fc2w16, CDIM * HID);

  // 2) BN + pack
  pack_bn_kernel<<<blocks((long)BSZ * NT * CDIM), TPB, 0, stream>>>(
      x, bn_g, bn_b, bn_m, bn_v, apack);

  // 3) QKV gemm  [NT x 512] @ [512 x 1536]
  gemm_wmma_kernel<0><<<dim3(1536 / 512, NT / 32, BSZ), TPB, 0, stream>>>(
      apack, qkvw16, nullptr, (void*)qkv16, CDIM, 3 * CDIM);

  // 4) V transpose pack, then fused attention
  vt_pack_kernel<<<blocks((long)BSZ * NH * HD * NT), TPB, 0, stream>>>(qkv16, vt16);
  attn_kernel<<<dim3(NT / 16, NH, BSZ), 32, 0, stream>>>(qkv16, vt16, o16);

  // 5) proj gemm (+bias)
  gemm_wmma_kernel<1><<<dim3(CDIM / 512, NT / 32, BSZ), TPB, 0, stream>>>(
      o16, projw16, proj_b, (void*)pout, CDIM, CDIM);

  // 6) residual + layer scale 1
  residual1_kernel<<<blocks((long)BSZ * CDIM * HWN), TPB, 0, stream>>>(x, pout, ls1, x1);

  // 7) BN -> dwconv -> fBN -> pack
  dwconv_kernel<<<blocks((long)BSZ * NT * CDIM), TPB, 0, stream>>>(
      x1, bn_g, bn_b, bn_m, bn_v, dw_w, fbn_g, fbn_b, fbn_m, fbn_v, apack);

  // 8) fc1 gemm (+bias, GELU)
  gemm_wmma_kernel<2><<<dim3(HID / 512, NT / 32, BSZ), TPB, 0, stream>>>(
      apack, fc1w16, fc1_b, (void*)qkv16, CDIM, HID);

  // 9) fc2 gemm (+bias)
  gemm_wmma_kernel<1><<<dim3(CDIM / 512, NT / 32, BSZ), TPB, 0, stream>>>(
      qkv16, fc2w16, fc2_b, (void*)pout, HID, CDIM);

  // 10) residual + layer scale 2 -> output
  final_kernel<<<blocks((long)BSZ * CDIM * HWN), TPB, 0, stream>>>(
      x1, pout, ls2, (float*)d_out);
}
